// MultiHeadAttentionLayer_180388627049
// MI455X (gfx1250) — matrix operand order repaired and checked by reading, not compiled, not run on hardware
//
#include <hip/hip_runtime.h>
#include <hip/hip_bf16.h>

typedef __attribute__((ext_vector_type(8)))  float  v8f;
typedef __attribute__((ext_vector_type(16))) __bf16 v16bf;
typedef __attribute__((ext_vector_type(8)))  __bf16 v8bf;
typedef unsigned short u16;

#define DEV __device__ __forceinline__

DEV u16 f2bf(float f) {
  unsigned u = __float_as_uint(f);
  u += 0x7fffu + ((u >> 16) & 1u);          // round-to-nearest-even
  return (u16)(u >> 16);
}
DEV float bf2f(u16 h) { return __uint_as_float(((unsigned)h) << 16); }

DEV v16bf mk16(v8bf a, v8bf b) {
  return __builtin_shufflevector(a, b, 0,1,2,3,4,5,6,7,8,9,10,11,12,13,14,15);
}
DEV v8f zero8() { v8f z = {0.f,0.f,0.f,0.f,0.f,0.f,0.f,0.f}; return z; }

DEV v8f wmma_bf16(v16bf a, v16bf b, v8f c) {
  // D = A(16x32 bf16) * B(32x16 bf16) + C(16x16 f32)
  return __builtin_amdgcn_wmma_f32_16x16x32_bf16(false, a, false, b, (short)0, c, false, false);
}

// ---- DPP butterfly reductions within each 16-lane group (one DPP row) ----
template <int CTRL>
DEV float dpp_mov(float x) {
  return __int_as_float(__builtin_amdgcn_mov_dpp(__float_as_int(x), CTRL, 0xF, 0xF, true));
}
DEV float redmax16(float x) {
  x = fmaxf(x, dpp_mov<0xB1>(x));    // quad_perm [1,0,3,2]  (xor 1)
  x = fmaxf(x, dpp_mov<0x4E>(x));    // quad_perm [2,3,0,1]  (xor 2)
  x = fmaxf(x, dpp_mov<0x141>(x));   // row_half_mirror      (combine quads)
  x = fmaxf(x, dpp_mov<0x140>(x));   // row_mirror           (combine 8-groups)
  return x;
}
DEV float redsum16(float x) {
  x += dpp_mov<0xB1>(x);
  x += dpp_mov<0x4E>(x);
  x += dpp_mov<0x141>(x);
  x += dpp_mov<0x140>(x);
  return x;
}

// ---------------------------------------------------------------- converts

__global__ void cvt_f32_bf16(const float* __restrict__ src, u16* __restrict__ dst, int n) {
  int i = blockIdx.x * 256 + threadIdx.x;
  if (i < n) dst[i] = f2bf(src[i]);
}

// Wt[n][k] = W[k][n]  (K-contiguous per output column -> contiguous B fragments)
__global__ void tcvt_f32_bf16(const float* __restrict__ W, u16* __restrict__ Wt) {
  int i = blockIdx.x * 256 + threadIdx.x;   // i = n*1024 + k
  int n = i >> 10, k = i & 1023;
  Wt[i] = f2bf(W[k * 1024 + n]);
}

// ---------------------------------------------------------------- GEMM
// C[M,N] = A[M,K](bf16) * Bt[N,K](bf16)^T + bias
// MODE 0: bf16 row-major out.  MODE 1: bf16 V^T per head [b,h,dh,S].  MODE 2: f32 out.
template <int MODE>
__global__ __launch_bounds__(256) void gemm_bf16(
    const u16* __restrict__ A, const u16* __restrict__ Bt,
    const float* __restrict__ bias, void* __restrict__ outp,
    int M, int N, int K) {
  const int lane = threadIdx.x & 31, wid = threadIdx.x >> 5;
  const int hi = lane >> 4, lo = lane & 15;
  const int m0 = blockIdx.x * 128 + (wid & 1) * 64;
  const int n0 = blockIdx.y * 256 + (wid >> 1) * 64;

  v8f acc[4][4];
#pragma unroll
  for (int i = 0; i < 4; ++i)
#pragma unroll
    for (int j = 0; j < 4; ++j) acc[i][j] = zero8();

  for (int k0 = 0; k0 < K; k0 += 32) {
    v16bf af[4], bf[4];
#pragma unroll
    for (int i = 0; i < 4; ++i) {
      const u16* p = A + (size_t)(m0 + 16 * i + lo) * K + k0 + hi * 8;
      af[i] = mk16(*(const v8bf*)p, *(const v8bf*)(p + 16));
      if (k0 + 32 < K) __builtin_prefetch(p + 32, 0, 3);
    }
#pragma unroll
    for (int j = 0; j < 4; ++j) {
      const u16* p = Bt + (size_t)(n0 + 16 * j + lo) * K + k0 + hi * 8;
      bf[j] = mk16(*(const v8bf*)p, *(const v8bf*)(p + 16));
      if (k0 + 32 < K) __builtin_prefetch(p + 32, 0, 3);
    }
#pragma unroll
    for (int i = 0; i < 4; ++i)
#pragma unroll
      for (int j = 0; j < 4; ++j) acc[i][j] = wmma_bf16(af[i], bf[j], acc[i][j]);
  }

#pragma unroll
  for (int i = 0; i < 4; ++i)
#pragma unroll
    for (int j = 0; j < 4; ++j) {
      const int col = n0 + 16 * j + lo;
      const float bcol = bias[col];
#pragma unroll
      for (int r = 0; r < 8; ++r) {
        const int row = m0 + 16 * i + r + 8 * hi;
        const float v = acc[i][j][r] + bcol;
        if (MODE == 0) {
          ((u16*)outp)[(size_t)row * N + col] = f2bf(v);
        } else if (MODE == 1) {
          const int bb = row >> 10, s = row & 1023;
          const int hh = col >> 6, d = col & 63;
          ((u16*)outp)[(((size_t)((bb * 16 + hh) * 64 + d)) << 10) + s] = f2bf(v);
        } else {
          ((float*)outp)[(size_t)row * N + col] = v;
        }
      }
    }
}

// ---------------------------------------------------------------- qrel
// qrel[b,h,q,r] = Q[b,q,h,:] . rel_k[r,:]   (r in [0,33))
__global__ void qrel_kernel(const u16* __restrict__ qbf, const float* __restrict__ relk,
                            float* __restrict__ qrel) {
  const int idx = blockIdx.x * 256 + threadIdx.x;
  const int TOT = 2 * 16 * 1024 * 33;
  if (idx >= TOT) return;
  const int r = idx % 33;
  const int q = (idx / 33) & 1023;
  const int h = (idx / (33 * 1024)) & 15;
  const int b = idx / (33 * 1024 * 16);
  const u16* qv = qbf + ((size_t)(b * 1024 + q)) * 1024 + h * 64;
  const float* rk = relk + r * 64;
  float acc = 0.f;
#pragma unroll 8
  for (int d = 0; d < 64; ++d) acc += bf2f(qv[d]) * rk[d];
  qrel[(((size_t)(b * 16 + h) * 1024) + q) * 33 + r] = acc;
}

// ---------------------------------------------------------------- attention

DEV v8f score_tile(const u16* kbase, int kt, int lo, int hi, v16bf aq0, v16bf aq1) {
  const u16* kp = kbase + (size_t)(kt * 16 + lo) * 1024 + hi * 8;
  v16bf bk0 = mk16(*(const v8bf*)kp, *(const v8bf*)(kp + 16));
  v16bf bk1 = mk16(*(const v8bf*)(kp + 32), *(const v8bf*)(kp + 48));
  v8f s = zero8();
  s = wmma_bf16(aq0, bk0, s);
  return wmma_bf16(aq1, bk1, s);
}

// Add Shaw bias with wave-uniform tile classification:
// tiles entirely past the +/-16 clip use register-resident clamp rows.
DEV void add_bias(const v8f& s, float* sv, const float* lqrel,
                  const float* blo, const float* bhi,
                  int kt, int qw, int q0, int lo, int hi) {
  const int dmaxT = kt * 16 + 15 - qw;      // max (k - q) in tile
  const int dminT = kt * 16 - (qw + 15);    // min (k - q) in tile
  if (dmaxT <= -16) {
#pragma unroll
    for (int r = 0; r < 8; ++r) sv[r] = (s[r] + blo[r]) * 0.125f;
  } else if (dminT >= 16) {
#pragma unroll
    for (int r = 0; r < 8; ++r) sv[r] = (s[r] + bhi[r]) * 0.125f;
  } else {
#pragma unroll
    for (int r = 0; r < 8; ++r) {
      const int qg = qw + r + 8 * hi;
      int dlt = kt * 16 + lo - qg;
      dlt = dlt < -16 ? -16 : (dlt > 16 ? 16 : dlt);
      sv[r] = (s[r] + lqrel[(qg - q0) * 33 + dlt + 16]) * 0.125f;
    }
  }
}

__global__ __launch_bounds__(256) void attn_kernel(
    const u16* __restrict__ qbf, const u16* __restrict__ kbf,
    const u16* __restrict__ vtbf, const float* __restrict__ qrel,
    const float* __restrict__ relv, u16* __restrict__ ax) {
  constexpr int S = 1024, D = 1024, H = 16, DH = 64;
  __shared__ float lqrel[128 * 33];
  __shared__ float lsrel[128 * 33];
  __shared__ __align__(16) u16 lp[8 * 512];   // per-wave 16x32 bf16 P tile

  const int tid = threadIdx.x;
  const int lane = tid & 31, wid = tid >> 5;
  const int hi = lane >> 4, lo = lane & 15;
  const int q0 = blockIdx.x * 128;
  const int h = blockIdx.y, b = blockIdx.z;
  const int qw = q0 + wid * 16;

  const size_t qrbase = ((size_t)(b * H + h) * S + q0) * 33;
  for (int i = tid; i < 128 * 33; i += 256) {
    lqrel[i] = qrel[qrbase + i];
    lsrel[i] = 0.f;
  }
  __syncthreads();

  // clamp-row biases held in registers (used by ~94% of tiles)
  float blo[8], bhi[8];
#pragma unroll
  for (int r = 0; r < 8; ++r) {
    const int ql = qw - q0 + r + 8 * hi;
    blo[r] = lqrel[ql * 33];
    bhi[r] = lqrel[ql * 33 + 32];
  }

  // Q A-fragments over dh (K = 0..31 and 32..63)
  const u16* qrow = qbf + ((size_t)(b * S + qw + lo)) * D + h * DH;
  const u16* p0 = qrow + hi * 8;
  const u16* p1 = qrow + 32 + hi * 8;
  const v16bf aq0 = mk16(*(const v8bf*)p0, *(const v8bf*)(p0 + 16));
  const v16bf aq1 = mk16(*(const v8bf*)p1, *(const v8bf*)(p1 + 16));

  const u16* kbase = kbf + ((size_t)b * S) * D + h * DH;
  const u16* vbase = vtbf + ((size_t)(b * H + h) * DH) * S;

  float m[8], l[8];
#pragma unroll
  for (int r = 0; r < 8; ++r) { m[r] = -1e30f; l[r] = 0.f; }

  // ---- pass 1: online softmax statistics (DPP reductions, no LDS traffic)
  for (int kt = 0; kt < 64; ++kt) {
    v8f s = score_tile(kbase, kt, lo, hi, aq0, aq1);
    float sv[8];
    add_bias(s, sv, lqrel, blo, bhi, kt, qw, q0, lo, hi);
#pragma unroll
    for (int r = 0; r < 8; ++r) {
      const float mx = redmax16(sv[r]);
      const float mn = fmaxf(m[r], mx);
      const float corr = __expf(m[r] - mn);
      const float ps = redsum16(__expf(sv[r] - mn));
      l[r] = l[r] * corr + ps;
      m[r] = mn;
    }
  }

  // ---- pass 2: recompute scores, accumulate O = P.V and srel bins
  v8f o[4];
#pragma unroll
  for (int j = 0; j < 4; ++j) o[j] = zero8();
  u16* lpw = lp + wid * 512;

  for (int ktp = 0; ktp < 32; ++ktp) {
#pragma unroll
    for (int sub = 0; sub < 2; ++sub) {
      const int kt = ktp * 2 + sub;
      v8f s = score_tile(kbase, kt, lo, hi, aq0, aq1);
      float sv[8];
      add_bias(s, sv, lqrel, blo, bhi, kt, qw, q0, lo, hi);
      const bool cLo = (kt * 16 + 15 - qw) <= -16;   // wave-uniform
      const bool cHi = (kt * 16 - (qw + 15)) >= 16;  // wave-uniform
#pragma unroll
      for (int r = 0; r < 8; ++r) {
        const int row = r + 8 * hi;
        const int qg = qw + row;
        const float p = __expf(sv[r] - m[r]);
        lpw[row * 32 + sub * 16 + lo] = f2bf(p);     // C-layout -> LDS
        if (cLo || cHi) {
          // saturated tile: one bin per row -> DPP row-sum + single atomic
          const float ps = redsum16(p);
          if (lo == 0) atomicAdd(&lsrel[(qg - q0) * 33 + (cLo ? 0 : 32)], ps);
        } else {
          int dlt = kt * 16 + lo - qg;
          dlt = dlt < -16 ? -16 : (dlt > 16 ? 16 : dlt);
          atomicAdd(&lsrel[(qg - q0) * 33 + dlt + 16], p);  // ds_add_f32
        }
      }
    }
    asm volatile("s_wait_dscnt 0" ::: "memory");     // wave-local DS ordering
    // reload P as an A-fragment (16 q x 32 keys)
    const u16* pp = lpw + lo * 32 + hi * 8;
    v16bf ap = mk16(*(const v8bf*)pp, *(const v8bf*)(pp + 16));
#pragma unroll
    for (int j = 0; j < 4; ++j) {
      const u16* vp = vbase + (size_t)(j * 16 + lo) * S + ktp * 32 + hi * 8;
      v16bf bvf = mk16(*(const v8bf*)vp, *(const v8bf*)(vp + 16));
      o[j] = wmma_bf16(ap, bvf, o[j]);
    }
  }
  asm volatile("s_wait_dscnt 0" ::: "memory");

  // ---- epilogue: add rel_v term (33 bins) and normalize by 1/l
  float linv[8];
#pragma unroll
  for (int r = 0; r < 8; ++r) linv[r] = 1.f / l[r];

#pragma unroll
  for (int j = 0; j < 4; ++j) {
    const int d = j * 16 + lo;
    float relpart[8];
#pragma unroll
    for (int r = 0; r < 8; ++r) relpart[r] = 0.f;
    for (int rr = 0; rr < 33; ++rr) {
      const float rvv = relv[rr * 64 + d];
#pragma unroll
      for (int r = 0; r < 8; ++r)
        relpart[r] += lsrel[(qw - q0 + r + 8 * hi) * 33 + rr] * rvv;
    }
#pragma unroll
    for (int r = 0; r < 8; ++r) {
      const int qg = qw + r + 8 * hi;
      const float val = (o[j][r] + relpart[r]) * linv[r];
      ax[((size_t)(b * S + qg)) * D + h * DH + d] = f2bf(val);
    }
  }
}

// ---------------------------------------------------------------- launch

extern "C" void kernel_launch(void* const* d_in, const int* in_sizes, int n_in,
                              void* d_out, int out_size, void* d_ws, size_t ws_size,
                              hipStream_t stream) {
  (void)in_sizes; (void)n_in; (void)out_size; (void)ws_size;
  const float* q_in = (const float*)d_in[0];
  const float* k_in = (const float*)d_in[1];
  const float* v_in = (const float*)d_in[2];
  const float* Wq = (const float*)d_in[3];  const float* bq = (const float*)d_in[4];
  const float* Wk = (const float*)d_in[5];  const float* bk = (const float*)d_in[6];
  const float* Wv = (const float*)d_in[7];  const float* bv = (const float*)d_in[8];
  const float* Wo = (const float*)d_in[9];  const float* bo = (const float*)d_in[10];
  const float* relk = (const float*)d_in[11];
  const float* relv = (const float*)d_in[12];

  char* ws = (char*)d_ws;
  const size_t MB = 1u << 20;
  u16* xq   = (u16*)(ws + 0 * MB);    // 4 MB each: bf16 activations
  u16* xk   = (u16*)(ws + 4 * MB);
  u16* xv   = (u16*)(ws + 8 * MB);
  u16* wqt  = (u16*)(ws + 12 * MB);   // 2 MB each: bf16 transposed weights
  u16* wkt  = (u16*)(ws + 14 * MB);
  u16* wvt  = (u16*)(ws + 16 * MB);
  u16* wot  = (u16*)(ws + 18 * MB);
  u16* qbf  = (u16*)(ws + 20 * MB);   // Q projection [B,S,D] bf16
  u16* kbf  = (u16*)(ws + 24 * MB);   // K projection [B,S,D] bf16
  u16* vtbf = (u16*)(ws + 28 * MB);   // V projection transposed [B,H,dh,S] bf16
  float* qr = (float*)(ws + 32 * MB); // qrel [B,H,S,33] f32 (4.2 MB)
  u16* ax   = (u16*)(ws + 37 * MB);   // attention output [B,S,D] bf16

  const int nBSD = 2 * 1024 * 1024;
  cvt_f32_bf16<<<nBSD / 256, 256, 0, stream>>>(q_in, xq, nBSD);
  cvt_f32_bf16<<<nBSD / 256, 256, 0, stream>>>(k_in, xk, nBSD);
  cvt_f32_bf16<<<nBSD / 256, 256, 0, stream>>>(v_in, xv, nBSD);
  tcvt_f32_bf16<<<4096, 256, 0, stream>>>(Wq, wqt);
  tcvt_f32_bf16<<<4096, 256, 0, stream>>>(Wk, wkt);
  tcvt_f32_bf16<<<4096, 256, 0, stream>>>(Wv, wvt);
  tcvt_f32_bf16<<<4096, 256, 0, stream>>>(Wo, wot);

  dim3 gg(16, 4);   // M/128 x N/256 for M=2048, N=1024
  gemm_bf16<0><<<gg, 256, 0, stream>>>(xq, wqt, bq, qbf, 2048, 1024, 1024);
  gemm_bf16<0><<<gg, 256, 0, stream>>>(xk, wkt, bk, kbf, 2048, 1024, 1024);
  gemm_bf16<1><<<gg, 256, 0, stream>>>(xv, wvt, bv, vtbf, 2048, 1024, 1024);

  const int TOT = 2 * 16 * 1024 * 33;
  qrel_kernel<<<(TOT + 255) / 256, 256, 0, stream>>>(qbf, relk, qr);

  attn_kernel<<<dim3(8, 16, 2), 256, 0, stream>>>(qbf, kbf, vtbf, qr, relv, ax);

  gemm_bf16<2><<<gg, 256, 0, stream>>>(ax, wot, bo, (float*)d_out, 2048, 1024, 1024);
}
